// MGNBlock_59803124629572
// MI455X (gfx1250) — matrix operand order repaired and checked by reading, not compiled, not run on hardware
//
#include <hip/hip_runtime.h>

// ---------------------------------------------------------------------------
// MGN block for MI455X (gfx1250): bf16 WMMA 16x16x32, weights staged in LDS
// in B-fragment layout, per-wave 16-row tiles, f32 accumulate + LayerNorm.
// Round 4: 8 waves/block (256 thr). Edge LDS = 160KB weights + 8x12.25KB
// per-wave scratch ~= 260KB < 320KB/WGP -> 2 waves/SIMD32 for latency hiding
// (4 waves/block left only 1 wave/SIMD: no inter-wave hiding of dscnt stalls).
// ---------------------------------------------------------------------------

typedef __attribute__((ext_vector_type(16))) __bf16 v16bf;
typedef __attribute__((ext_vector_type(2)))  __bf16 v2bf;
typedef __attribute__((ext_vector_type(8)))  float  v8f;

#define WAVES_PER_BLOCK 8
#define TPB (WAVES_PER_BLOCK * 32)
// per-wave LDS: hb(16x128 bf16)=4096 + hf(16x128 f32)=8192 + stats 128 + meta 64 + mask 64
#define PER_WAVE_BYTES 12544

__device__ __forceinline__ unsigned packpair(float a, float b) {
  v2bf t;
  t.x = (__bf16)a;
  t.y = (__bf16)b;
  return __builtin_bit_cast(unsigned, t);
}

__device__ __forceinline__ v16bf pack16(const float* f) {
  v16bf v;
#pragma unroll
  for (int i = 0; i < 16; ++i) v[i] = (__bf16)f[i];
  return v;
}

// B-fragment fetch: wb laid out as [kstep][tile][lane][8 packed-pairs]
__device__ __forceinline__ v16bf loadB(const unsigned* wb, int ks, int t, int lane) {
  const unsigned* p = wb + (size_t)(((((ks << 3) + t) << 5) + lane) * 8);
  union { v16bf v; uint4 q[2]; } u;
  u.q[0] = *(const uint4*)p;
  u.q[1] = *(const uint4*)(p + 4);
  return u.v;
}

// A-fragment from LDS bf16 [16][128]: lane l -> row l&15, halves per ISA layout
__device__ __forceinline__ v16bf loadA16(const __bf16* hb, int m, int kb, int half) {
  union { v16bf v; uint4 q[2]; } u;
  u.q[0] = *(const uint4*)(hb + m * 128 + kb + half * 8);
  u.q[1] = *(const uint4*)(hb + m * 128 + kb + 16 + half * 8);
  return u.v;
}

// Stage f32 weights [K][128] -> LDS bf16 pairs in B-fragment order.
__device__ __forceinline__ void stageW(unsigned* dst, const float* __restrict__ W, int K,
                                       int tid, int nth) {
  const int total = (K >> 1) * 128;
  for (int idx = tid; idx < total; idx += nth) {
    const int k2 = idx >> 7;
    const int nn = idx & 127;
    const int k = k2 << 1;
    const unsigned pr = packpair(W[(size_t)k * 128 + nn], W[(size_t)(k + 1) * 128 + nn]);
    const int kstep = k2 >> 4;
    const int jj = k2 & 15;
    const int khalf = (jj >> 2) & 1;
    const int j = (jj & 3) + ((jj >> 3) << 2);
    const int t = nn >> 4;
    const int nl = nn & 15;
    const int ln = nl + (khalf << 4);
    dst[(size_t)(((((kstep << 3) + t) << 5) + ln) * 8 + j)] = pr;
  }
}

__device__ __forceinline__ float silu(float x) {
  return x * __builtin_amdgcn_rcpf(1.f + __expf(-x));
}

#define WMMA_BF16(a, b, c) \
  __builtin_amdgcn_wmma_f32_16x16x32_bf16(false, (a), false, (b), (short)0, (c), false, false)

// One k-step of a 16x(K)x128 GEMM tile: batch-load 8 B fragments, then 8 WMMAs.
__device__ __forceinline__ void gemm_step(v8f acc[8], v16bf a, const unsigned* wb,
                                          int ks, int lane) {
  v16bf bfr[8];
#pragma unroll
  for (int t = 0; t < 8; ++t) bfr[t] = loadB(wb, ks, t, lane);
#pragma unroll
  for (int t = 0; t < 8; ++t) acc[t] = WMMA_BF16(a, bfr[t], acc[t]);
}

// ---------------------------------------------------------------------------
// Edge kernel
// ---------------------------------------------------------------------------
__global__ void __launch_bounds__(TPB) mgn_edge_kernel(
    const float* __restrict__ V, const float* __restrict__ E, const int* __restrict__ edges,
    const float* __restrict__ W0, const float* __restrict__ b0,
    const float* __restrict__ W1, const float* __restrict__ b1,
    const float* __restrict__ W2, const float* __restrict__ b2,
    const float* __restrict__ g, const float* __restrict__ be,
    float* __restrict__ Eout, float* __restrict__ sums, float* __restrict__ counts,
    int n, int ne)
{
  extern __shared__ char smem[];
  unsigned* wB0 = (unsigned*)smem;
  unsigned* wB1 = wB0 + 384 * 64;
  unsigned* wB2 = wB1 + 128 * 64;
  char* pw = smem + 640 * 64 * 4;

  const int wave = threadIdx.x >> 5;
  const int lane = threadIdx.x & 31;
  char* wb = pw + wave * PER_WAVE_BYTES;
  __bf16* hb   = (__bf16*)wb;
  float* hf    = (float*)(wb + 4096);
  float* stats = (float*)(wb + 4096 + 8192);
  int*   meta  = (int*)(wb + 4096 + 8192 + 128);
  float* fmask = (float*)(wb + 4096 + 8192 + 128 + 64);

  stageW(wB0, W0, 384, threadIdx.x, TPB);
  stageW(wB1, W1, 128, threadIdx.x, TPB);
  stageW(wB2, W2, 128, threadIdx.x, TPB);
  __syncthreads();

  const int m = lane & 15;
  const int half = lane >> 4;
  const int nl = lane & 15;
  const int ntiles = (ne + 15) >> 4;

  for (int tile = blockIdx.x * WAVES_PER_BLOCK + wave; tile < ntiles;
       tile += gridDim.x * WAVES_PER_BLOCK) {
    const int eidx = tile * 16 + m;
    const int rowok = (eidx < ne);
    const int ec = rowok ? eidx : (ne - 1);
    const int s0 = edges[2 * ec], r0 = edges[2 * ec + 1];
    const float mk = (s0 >= 0 && r0 >= 0 && rowok) ? 1.f : 0.f;
    const int ss = min(max(s0, 0), n - 1);
    const int rr = min(max(r0, 0), n - 1);
    if (half == 0) { meta[m] = rr; fmask[m] = mk; }
    const float* pS = V + (size_t)ss * 128;
    const float* pR = V + (size_t)rr * 128;
    const float* pE = E + (size_t)ec * 128;

    v8f acc[8] = {};
#pragma unroll
    for (int ks = 0; ks < 12; ++ks) {
      const int kb = ks * 32;
      const int reg = kb >> 7;
      const int off = kb & 127;
      const float* base = (reg == 0) ? pS : ((reg == 1) ? pR : pE);
      const float* p0 = base + off + half * 8;
      float f[16];
#pragma unroll
      for (int i = 0; i < 8; ++i) { f[i] = p0[i]; f[8 + i] = p0[16 + i]; }
      v16bf a = pack16(f);
      gemm_step(acc, a, wB0, ks, lane);
    }
#pragma unroll
    for (int t = 0; t < 8; ++t) {
      const float bias = b0[t * 16 + nl];
#pragma unroll
      for (int j = 0; j < 8; ++j)
        hb[(j + 8 * half) * 128 + t * 16 + nl] = (__bf16)silu(acc[t][j] + bias);
    }

    v8f acc1[8] = {};
#pragma unroll
    for (int ks = 0; ks < 4; ++ks) {
      v16bf a = loadA16(hb, m, ks * 32, half);
      gemm_step(acc1, a, wB1, ks, lane);
    }
#pragma unroll
    for (int t = 0; t < 8; ++t) {
      const float bias = b1[t * 16 + nl];
#pragma unroll
      for (int j = 0; j < 8; ++j)
        hb[(j + 8 * half) * 128 + t * 16 + nl] = (__bf16)silu(acc1[t][j] + bias);
    }

    v8f acc2[8] = {};
#pragma unroll
    for (int ks = 0; ks < 4; ++ks) {
      v16bf a = loadA16(hb, m, ks * 32, half);
      gemm_step(acc2, a, wB2, ks, lane);
    }
#pragma unroll
    for (int t = 0; t < 8; ++t) {
      const float bias = b2[t * 16 + nl];
#pragma unroll
      for (int j = 0; j < 8; ++j)
        hf[(j + 8 * half) * 128 + t * 16 + nl] = acc2[t][j] + bias;
    }

    if (half == 0) {
      const float* hr = hf + m * 128;
      float s = 0.f, s2 = 0.f;
#pragma unroll
      for (int i = 0; i < 32; ++i) {
        float4 v4 = *(const float4*)(hr + i * 4);
        s  += v4.x + v4.y + v4.z + v4.w;
        s2 += v4.x * v4.x + v4.y * v4.y + v4.z * v4.z + v4.w * v4.w;
      }
      const float mu = s * 0.0078125f;
      const float var = s2 * 0.0078125f - mu * mu;
      stats[m] = mu;
      stats[16 + m] = __frsqrt_rn(var + 1e-5f);
    }

    const int c0 = lane * 4;
    const float4 gg = *(const float4*)(g + c0);
    const float4 bb = *(const float4*)(be + c0);
    for (int r2 = 0; r2 < 16; ++r2) {
      const int er = tile * 16 + r2;
      if (er < ne) {
        const float mu = stats[r2], rs = stats[16 + r2];
        const float mk2 = fmask[r2];
        const float4 x = *(const float4*)(hf + r2 * 128 + c0);
        float4 y;
        y.x = (x.x - mu) * rs * gg.x + bb.x;
        y.y = (x.y - mu) * rs * gg.y + bb.y;
        y.z = (x.z - mu) * rs * gg.z + bb.z;
        y.w = (x.w - mu) * rs * gg.w + bb.w;
        const size_t eb = (size_t)er * 128 + c0;
        const float4 eo = *(const float4*)(E + eb);
        float4 en;
        en.x = eo.x + y.x * mk2;
        en.y = eo.y + y.y * mk2;
        en.z = eo.z + y.z * mk2;
        en.w = eo.w + y.w * mk2;
        *(float4*)(Eout + eb) = en;
        if (mk2 != 0.f) {
          float* sp = sums + (size_t)meta[r2] * 128 + c0;
          atomicAdd(sp + 0, en.x);
          atomicAdd(sp + 1, en.y);
          atomicAdd(sp + 2, en.z);
          atomicAdd(sp + 3, en.w);
        }
      }
    }
    if (half == 0 && rowok) atomicAdd(counts + meta[m], 1.f);
  }
}

// ---------------------------------------------------------------------------
// Node kernel
// ---------------------------------------------------------------------------
__global__ void __launch_bounds__(TPB) mgn_node_kernel(
    const float* __restrict__ V, const float* __restrict__ sums, const float* __restrict__ counts,
    const float* __restrict__ W0, const float* __restrict__ b0,
    const float* __restrict__ W1, const float* __restrict__ b1,
    const float* __restrict__ W2, const float* __restrict__ b2,
    const float* __restrict__ g, const float* __restrict__ be,
    float* __restrict__ Vout, int nn)
{
  extern __shared__ char smem[];
  unsigned* wB0 = (unsigned*)smem;
  unsigned* wB1 = wB0 + 256 * 64;
  unsigned* wB2 = wB1 + 128 * 64;
  char* pw = smem + 512 * 64 * 4;

  const int wave = threadIdx.x >> 5;
  const int lane = threadIdx.x & 31;
  char* wb = pw + wave * PER_WAVE_BYTES;
  __bf16* hb   = (__bf16*)wb;
  float* hf    = (float*)(wb + 4096);
  float* stats = (float*)(wb + 4096 + 8192);

  stageW(wB0, W0, 256, threadIdx.x, TPB);
  stageW(wB1, W1, 128, threadIdx.x, TPB);
  stageW(wB2, W2, 128, threadIdx.x, TPB);
  __syncthreads();

  const int m = lane & 15;
  const int half = lane >> 4;
  const int nl = lane & 15;
  const int ntiles = (nn + 15) >> 4;

  for (int tile = blockIdx.x * WAVES_PER_BLOCK + wave; tile < ntiles;
       tile += gridDim.x * WAVES_PER_BLOCK) {
    const int node = tile * 16 + m;
    const int nc = (node < nn) ? node : (nn - 1);
    const float* pV = V + (size_t)nc * 128;
    const float* pA = sums + (size_t)nc * 128;
    const float cnt = counts[nc];
    const float inv = __builtin_amdgcn_rcpf(fmaxf(cnt, 1.f));

    v8f acc[8] = {};
#pragma unroll
    for (int ks = 0; ks < 8; ++ks) {
      const int kb = ks * 32;
      const int off = kb & 127;
      const bool isagg = (kb >= 128);
      const float* p0 = (isagg ? pA : pV) + off + half * 8;
      const float sc = isagg ? inv : 1.f;
      float f[16];
#pragma unroll
      for (int i = 0; i < 8; ++i) { f[i] = p0[i] * sc; f[8 + i] = p0[16 + i] * sc; }
      v16bf a = pack16(f);
      gemm_step(acc, a, wB0, ks, lane);
    }
#pragma unroll
    for (int t = 0; t < 8; ++t) {
      const float bias = b0[t * 16 + nl];
#pragma unroll
      for (int j = 0; j < 8; ++j)
        hb[(j + 8 * half) * 128 + t * 16 + nl] = (__bf16)silu(acc[t][j] + bias);
    }

    v8f acc1[8] = {};
#pragma unroll
    for (int ks = 0; ks < 4; ++ks) {
      v16bf a = loadA16(hb, m, ks * 32, half);
      gemm_step(acc1, a, wB1, ks, lane);
    }
#pragma unroll
    for (int t = 0; t < 8; ++t) {
      const float bias = b1[t * 16 + nl];
#pragma unroll
      for (int j = 0; j < 8; ++j)
        hb[(j + 8 * half) * 128 + t * 16 + nl] = (__bf16)silu(acc1[t][j] + bias);
    }

    v8f acc2[8] = {};
#pragma unroll
    for (int ks = 0; ks < 4; ++ks) {
      v16bf a = loadA16(hb, m, ks * 32, half);
      gemm_step(acc2, a, wB2, ks, lane);
    }
#pragma unroll
    for (int t = 0; t < 8; ++t) {
      const float bias = b2[t * 16 + nl];
#pragma unroll
      for (int j = 0; j < 8; ++j)
        hf[(j + 8 * half) * 128 + t * 16 + nl] = acc2[t][j] + bias;
    }

    if (half == 0) {
      const float* hr = hf + m * 128;
      float s = 0.f, s2 = 0.f;
#pragma unroll
      for (int i = 0; i < 32; ++i) {
        float4 v4 = *(const float4*)(hr + i * 4);
        s  += v4.x + v4.y + v4.z + v4.w;
        s2 += v4.x * v4.x + v4.y * v4.y + v4.z * v4.z + v4.w * v4.w;
      }
      const float mu = s * 0.0078125f;
      const float var = s2 * 0.0078125f - mu * mu;
      stats[m] = mu;
      stats[16 + m] = __frsqrt_rn(var + 1e-5f);
    }

    const int c0 = lane * 4;
    const float4 gg = *(const float4*)(g + c0);
    const float4 bb = *(const float4*)(be + c0);
    for (int r2 = 0; r2 < 16; ++r2) {
      const int vr = tile * 16 + r2;
      if (vr < nn) {
        const float mu = stats[r2], rs = stats[16 + r2];
        const float4 x = *(const float4*)(hf + r2 * 128 + c0);
        const size_t vb = (size_t)vr * 128 + c0;
        const float4 vo = *(const float4*)(V + vb);
        float4 y;
        y.x = vo.x + (x.x - mu) * rs * gg.x + bb.x;
        y.y = vo.y + (x.y - mu) * rs * gg.y + bb.y;
        y.z = vo.z + (x.z - mu) * rs * gg.z + bb.z;
        y.w = vo.w + (x.w - mu) * rs * gg.w + bb.w;
        *(float4*)(Vout + vb) = y;
      }
    }
  }
}

// ---------------------------------------------------------------------------
extern "C" void kernel_launch(void* const* d_in, const int* in_sizes, int n_in,
                              void* d_out, int out_size, void* d_ws, size_t ws_size,
                              hipStream_t stream) {
  const float* V     = (const float*)d_in[0];
  const float* E     = (const float*)d_in[1];
  const int*   edges = (const int*)d_in[2];
  const float* feW0 = (const float*)d_in[3],  *feb0 = (const float*)d_in[4];
  const float* feW1 = (const float*)d_in[5],  *feb1 = (const float*)d_in[6];
  const float* feW2 = (const float*)d_in[7],  *feb2 = (const float*)d_in[8];
  const float* feg  = (const float*)d_in[9],  *febe = (const float*)d_in[10];
  const float* fnW0 = (const float*)d_in[11], *fnb0 = (const float*)d_in[12];
  const float* fnW1 = (const float*)d_in[13], *fnb1 = (const float*)d_in[14];
  const float* fnW2 = (const float*)d_in[15], *fnb2 = (const float*)d_in[16];
  const float* fng  = (const float*)d_in[17], *fnbe = (const float*)d_in[18];

  const int n  = in_sizes[0] / 128;   // nodes
  const int ne = in_sizes[1] / 128;   // edges

  float* Vout   = (float*)d_out;
  float* Eout   = Vout + (size_t)n * 128;
  float* sums   = (float*)d_ws;
  float* counts = sums + (size_t)n * 128;

  hipMemsetAsync(d_ws, 0, ((size_t)n * 129) * sizeof(float), stream);

  const size_t shE = (size_t)640 * 64 * 4 + WAVES_PER_BLOCK * PER_WAVE_BYTES; // ~260 KB
  const size_t shN = (size_t)512 * 64 * 4 + WAVES_PER_BLOCK * PER_WAVE_BYTES; // ~226 KB
  hipFuncSetAttribute((const void*)mgn_edge_kernel,
                      hipFuncAttributeMaxDynamicSharedMemorySize, (int)shE);
  hipFuncSetAttribute((const void*)mgn_node_kernel,
                      hipFuncAttributeMaxDynamicSharedMemorySize, (int)shN);

  const int ntE = (ne + 15) / 16;
  int blocksE = (ntE + WAVES_PER_BLOCK - 1) / WAVES_PER_BLOCK;
  if (blocksE > 1024) blocksE = 1024;
  mgn_edge_kernel<<<blocksE, TPB, shE, stream>>>(
      V, E, edges, feW0, feb0, feW1, feb1, feW2, feb2, feg, febe,
      Eout, sums, counts, n, ne);

  const int ntN = (n + 15) / 16;
  int blocksN = (ntN + WAVES_PER_BLOCK - 1) / WAVES_PER_BLOCK;
  if (blocksN > 1024) blocksN = 1024;
  mgn_node_kernel<<<blocksN, TPB, shN, stream>>>(
      V, sums, counts, fnW0, fnb0, fnW1, fnb1, fnW2, fnb2, fng, fnbe,
      Vout, n);

  (void)n_in; (void)out_size; (void)ws_size;
}
